// up_DKN_20074677142173
// MI455X (gfx1250) — matrix operand order repaired and checked by reading, not compiled, not run on hardware
//
#include <hip/hip_runtime.h>
#include <stdint.h>

#define DEV_INLINE __device__ __forceinline__

constexpr int Bn  = 4;
constexpr int Cc  = 64;
constexpr int Hh  = 240;
constexpr int Wd  = 320;
constexpr int HWc = Hh * Wd;            // 76800
constexpr int NPIX = Bn * HWc;          // 307200
constexpr int NKB  = 36;                // K blocks of 32 (K = 9*128 = 1152)
constexpr float EPSV  = 1e-5f;
constexpr float SLOPE = 0.2f;

typedef __attribute__((ext_vector_type(16))) __bf16        v16bf;
typedef __attribute__((ext_vector_type(8)))  float         v8f;
typedef __attribute__((ext_vector_type(8)))  unsigned int  v8u;

union U32x8 {                            // 32-byte fragment, uint4-pair view
  uint4 q[2];
  v8u   v;
};

DEV_INLINE unsigned short f2bf(float f) {
  unsigned int u = __builtin_bit_cast(unsigned int, f);
  unsigned int r = u + 0x7FFFu + ((u >> 16) & 1u);   // round-to-nearest-even
  return (unsigned short)(r >> 16);
}

DEV_INLINE v8f wmma_bf16(v8u a, v8u b, v8f c) {
  return __builtin_amdgcn_wmma_f32_16x16x32_bf16(
      false, __builtin_bit_cast(v16bf, a),
      false, __builtin_bit_cast(v16bf, b),
      (short)0, c, false, false);
}

// ---------------------------------------------------------------------------
// Kernel 1: build cat = [df(64) | feature(64)] as bf16, channels-last
//   cat[(b*H+h)*W+w][c], c in 0..127, 256 B per pixel row.
//   One block = 32 pixels; feature transposed through LDS so both the
//   strided NCHW reads and the channels-last writes are coalesced.
// ---------------------------------------------------------------------------
__global__ __launch_bounds__(256) void prep_cat(
    const float* __restrict__ depth, const float* __restrict__ feature,
    const float* __restrict__ w1, const float* __restrict__ g1,
    const float* __restrict__ b1, const float* __restrict__ m1,
    const float* __restrict__ v1, unsigned short* __restrict__ cat) {
  __shared__ float lfeat[64][33];
  __shared__ float ldep[32];
  const int tid = threadIdx.x;
  const int p0  = blockIdx.x * 32;          // HW % 32 == 0 -> no batch straddle
  const int b   = p0 / HWc;
  const int hw0 = p0 - b * HWc;

  const int cr = tid >> 5;                  // 0..7
  const int pr = tid & 31;
  const float* fb = feature + (size_t)b * 64 * HWc + hw0;
#pragma unroll
  for (int k = 0; k < 8; ++k) {
    const int cc = cr + k * 8;
    lfeat[cc][pr] = fb[(size_t)cc * HWc + pr];
  }
  if (tid < 32) ldep[tid] = depth[(size_t)b * HWc + hw0 + tid];
  __syncthreads();

  const int wp = tid >> 3;
  const int c0 = (tid & 7) * 16;
  const float d = ldep[wp];
  unsigned int outw[8];
#pragma unroll
  for (int e = 0; e < 8; ++e) {
    const int ch0 = c0 + e * 2;
    const int ch1 = ch0 + 1;
    float x0, x1;
    if (ch0 < 64) {                         // depth-guided branch (df)
      float inv = g1[ch0] * rsqrtf(v1[ch0] + EPSV);
      float xs  = d * (w1[ch0] * inv) + (b1[ch0] - m1[ch0] * inv);
      x0 = xs >= 0.f ? xs : SLOPE * xs;
      inv = g1[ch1] * rsqrtf(v1[ch1] + EPSV);
      xs  = d * (w1[ch1] * inv) + (b1[ch1] - m1[ch1] * inv);
      x1 = xs >= 0.f ? xs : SLOPE * xs;
    } else {                                // feature passthrough
      x0 = lfeat[ch0 - 64][wp];
      x1 = lfeat[ch1 - 64][wp];
    }
    outw[e] = (unsigned int)f2bf(x0) | ((unsigned int)f2bf(x1) << 16);
  }
  uint4* dst = (uint4*)(cat + (size_t)(p0 + wp) * 128 + c0);
  dst[0] = make_uint4(outw[0], outw[1], outw[2], outw[3]);
  dst[1] = make_uint4(outw[4], outw[5], outw[6], outw[7]);
}

// ---------------------------------------------------------------------------
// Kernel 2: fold BN scale into conv weights, pre-swizzle into the CDNA5
//   16-bit A-matrix 16x32 lane layout. Apack[mtile][kb][lane][e] bf16,
//   mtile in {0,1} (M rows 0-15 / 16-31, rows 27-31 zero), kb in 0..35.
//   K ordering: K = kpos*128 + c, kpos = ky*3+kx. Also emits the 27 biases
//   and zero-fills the 256 B zero-row used for branchless conv padding.
// ---------------------------------------------------------------------------
__global__ __launch_bounds__(256) void repack_w(
    const float* __restrict__ ww, const float* __restrict__ gw,
    const float* __restrict__ bw, const float* __restrict__ mw,
    const float* __restrict__ vw, const float* __restrict__ wo,
    const float* __restrict__ go, const float* __restrict__ bo,
    const float* __restrict__ mo, const float* __restrict__ vo,
    unsigned short* __restrict__ Apack, float* __restrict__ bias,
    unsigned int* __restrict__ zrow) {
  const int t = blockIdx.x * 256 + threadIdx.x;   // over 2*36*32*16 = 36864
  if (t < 2 * NKB * 32 * 16) {
    const int e     = t & 15;
    const int lane  = (t >> 4) & 31;
    const int kb    = (t >> 9) % NKB;
    const int mtile = t / (NKB * 512);
    const int M     = mtile * 16 + (lane & 15);
    const int khalf = lane >> 4;
    // A 16x32 bf16 layout: lanes 0-15: e0-7 -> K0-7, e8-15 -> K16-23;
    //                      lanes 16-31: e0-7 -> K8-15, e8-15 -> K24-31.
    const int klocal = (e < 8) ? (khalf * 8 + e) : (16 + khalf * 8 + (e - 8));
    const int K    = kb * 32 + klocal;
    const int kpos = K >> 7;
    const int cch  = K & 127;
    const int ky = kpos / 3, kx = kpos % 3;
    float val = 0.f;
    if (M < 9) {
      const float inv = gw[M] * rsqrtf(vw[M] + EPSV);
      val = ww[((M * 128 + cch) * 3 + ky) * 3 + kx] * inv;
    } else if (M < 27) {
      const int o = M - 9;
      const float inv = go[o] * rsqrtf(vo[o] + EPSV);
      val = wo[((o * 128 + cch) * 3 + ky) * 3 + kx] * inv;
    }
    Apack[t] = f2bf(val);
  }
  if (t < 27) {
    float bv;
    if (t < 9) {
      const float inv = gw[t] * rsqrtf(vw[t] + EPSV);
      bv = bw[t] - mw[t] * inv;
    } else {
      const int o = t - 9;
      const float inv = go[o] * rsqrtf(vo[o] + EPSV);
      bv = bo[o] - mo[o] * inv;
    }
    bias[t] = bv;
  }
  if (t < 64) zrow[t] = 0u;                 // 256 B zero pixel-row
}

// ---------------------------------------------------------------------------
// Kernel 3: im2col GEMM via v_wmma_f32_16x16x32_bf16.
//   M=32 (27 used), K=1152, N=NPIX. Block = 8 waves; each wave owns TWO
//   16-pixel column tiles (32 consecutive pixels, always within one image
//   row since W%32==0) and both M-tiles -> 4 accumulators, so every A
//   fragment is reused for two B fragments. A (73728 B) is staged in LDS
//   and read back as ds_load_b128. Conv zero-padding is branchless: invalid
//   rows redirect the B-load pointer to a 256 B zero row (L0-resident).
// ---------------------------------------------------------------------------
__global__ __launch_bounds__(256) void gemm_wmma(
    const unsigned short* __restrict__ cat,
    const unsigned short* __restrict__ Apack,
    const unsigned short* __restrict__ zrow,
    float* __restrict__ gacc) {
  extern __shared__ uint4 sA4[];            // 4608 uint4 = 73728 B
  const int tid = threadIdx.x;
  {
    const uint4* src = (const uint4*)Apack;
#pragma unroll
    for (int i = 0; i < 18; ++i)            // 18*256 = 4608 uint4
      sA4[tid + i * 256] = src[tid + i * 256];
  }
  __syncthreads();

  const int wid   = tid >> 5;
  const int lane  = tid & 31;
  const int n     = lane & 15;              // column within tile
  const int khalf = lane >> 4;              // B: lanes 0-15 K0-7,16-23; 16-31 K8-15,24-31
  const int p0 = blockIdx.x * 256 + wid * 32 + n;   // tile0 pixel
  const int b  = p0 / HWc;
  const int hw = p0 - b * HWc;
  const int h  = hw / Wd;
  const int w0 = hw - h * Wd;               // tile1 pixel is at w0+16, same row

  const v8f vzero = {0.f, 0.f, 0.f, 0.f, 0.f, 0.f, 0.f, 0.f};
  v8f acc00 = vzero, acc01 = vzero, acc10 = vzero, acc11 = vzero;

  for (int kpos = 0; kpos < 9; ++kpos) {
    const int h2 = h + (kpos / 3) - 1;
    const int dx = (kpos % 3) - 1;
    const bool hv = (unsigned)h2 < (unsigned)Hh;
    const int w2a = w0 + dx;
    const int w2b = w0 + 16 + dx;
    const long long base = (long long)(b * Hh + h2) * Wd;
    const unsigned short* rowa =
        (hv && (unsigned)w2a < (unsigned)Wd) ? cat + (base + w2a) * 128 : zrow;
    const unsigned short* rowb =
        (hv && (unsigned)w2b < (unsigned)Wd) ? cat + (base + w2b) * 128 : zrow;
#pragma unroll
    for (int cb = 0; cb < 4; ++cb) {
      const int kb = kpos * 4 + cb;
      const int co = cb * 32 + khalf * 8;   // channel offset: K 8h..8h+7, 16+8h..16+8h+7
      U32x8 a0, a1, bfa, bfb;
      const uint4* ap0 = sA4 + (size_t)(kb * 32 + lane) * 2;
      const uint4* ap1 = sA4 + (size_t)((NKB + kb) * 32 + lane) * 2;
      a0.q[0] = ap0[0];  a0.q[1] = ap0[1];
      a1.q[0] = ap1[0];  a1.q[1] = ap1[1];
      const uint4* bpa = (const uint4*)(rowa + co);
      const uint4* bpb = (const uint4*)(rowb + co);
      bfa.q[0] = bpa[0]; bfa.q[1] = bpa[2];
      bfb.q[0] = bpb[0]; bfb.q[1] = bpb[2];
      acc00 = wmma_bf16(a0.v, bfa.v, acc00);
      acc01 = wmma_bf16(a0.v, bfb.v, acc01);
      acc10 = wmma_bf16(a1.v, bfa.v, acc10);
      acc11 = wmma_bf16(a1.v, bfb.v, acc11);
    }
  }

  // D layout: lanes 0-15 -> M rows 0-7, lanes 16-31 -> rows 8-15 (per tile)
  float* g0 = gacc + (long long)p0 * 32 + khalf * 8;
  *(float4*)(g0 + 0)  = make_float4(acc00[0], acc00[1], acc00[2], acc00[3]);
  *(float4*)(g0 + 4)  = make_float4(acc00[4], acc00[5], acc00[6], acc00[7]);
  *(float4*)(g0 + 16) = make_float4(acc10[0], acc10[1], acc10[2], acc10[3]);
  *(float4*)(g0 + 20) = make_float4(acc10[4], acc10[5], acc10[6], acc10[7]);
  float* g1p = gacc + (long long)(p0 + 16) * 32 + khalf * 8;
  *(float4*)(g1p + 0)  = make_float4(acc01[0], acc01[1], acc01[2], acc01[3]);
  *(float4*)(g1p + 4)  = make_float4(acc01[4], acc01[5], acc01[6], acc01[7]);
  *(float4*)(g1p + 16) = make_float4(acc11[0], acc11[1], acc11[2], acc11[3]);
  *(float4*)(g1p + 20) = make_float4(acc11[4], acc11[5], acc11[6], acc11[7]);
}

// ---------------------------------------------------------------------------
// Kernel 4: per-pixel epilogue: bias + sigmoid + zero-mean weights, then
//   9 deformable taps, each a 4-corner bilinear gather from depth with the
//   reference's double validity rule (15x15 patch window AND image bounds).
// ---------------------------------------------------------------------------
DEV_INLINE float gather_d(const float* __restrict__ dp, int h, int w,
                          float xf, float yf) {
  if (xf < 0.f || xf > 14.f || yf < 0.f || yf > 14.f) return 0.f;
  const int xi = (int)xf;                   // xf in [0,14], already floored
  const int yi = (int)yf;
  const int hh = h - 7 + yi;
  const int wc = w - 7 + xi;
  if ((unsigned)hh >= (unsigned)Hh || (unsigned)wc >= (unsigned)Wd) return 0.f;
  return dp[hh * Wd + wc];
}

__global__ __launch_bounds__(256) void dkn_sample(
    const float* __restrict__ gacc, const float* __restrict__ bias,
    const float* __restrict__ depth, float* __restrict__ out) {
  const int p = blockIdx.x * 256 + threadIdx.x;
  if (p >= NPIX) return;
  const int b  = p / HWc;
  const int hw = p - b * HWc;
  const int h  = hw / Wd;
  const int w  = hw - h * Wd;
  const float* gp = gacc + (long long)p * 32;

  float wgt[9], off[18];
  float s = 0.f;
#pragma unroll
  for (int i = 0; i < 9; ++i) {
    const float x = gp[i] + bias[i];
    const float sg = 1.f / (1.f + expf(-x));
    wgt[i] = sg;
    s += sg;
  }
  s *= (1.f / 9.f);
#pragma unroll
  for (int i = 0; i < 9; ++i) wgt[i] -= s;
#pragma unroll
  for (int i = 0; i < 18; ++i) off[i] = gp[9 + i] + bias[9 + i];

  const float* dp = depth + (size_t)b * HWc;
  float acc = 0.f;
#pragma unroll
  for (int i = 0; i < 3; ++i) {
#pragma unroll
    for (int j = 0; j < 3; ++j) {
      const int t = i * 3 + j;
      // grid: gx[i][j] = 7+i (x), gy[i][j] = 7+j (y); ix = (g+off)*14/15
      const float ix = ((float)(7 + i) + off[t * 2 + 0]) * (14.f / 15.f);
      const float iy = ((float)(7 + j) + off[t * 2 + 1]) * (14.f / 15.f);
      const float x0 = floorf(ix), y0 = floorf(iy);
      const float wx1 = ix - x0, wy1 = iy - y0;
      const float v00 = gather_d(dp, h, w, x0, y0);
      const float v10 = gather_d(dp, h, w, x0 + 1.f, y0);
      const float v01 = gather_d(dp, h, w, x0, y0 + 1.f);
      const float v11 = gather_d(dp, h, w, x0 + 1.f, y0 + 1.f);
      const float samp = v00 * (1.f - wx1) * (1.f - wy1) + v10 * wx1 * (1.f - wy1)
                       + v01 * (1.f - wx1) * wy1 + v11 * wx1 * wy1;
      acc += samp * wgt[t];
    }
  }
  out[p] = acc + dp[hw];
}

// ---------------------------------------------------------------------------
extern "C" void kernel_launch(void* const* d_in, const int* in_sizes, int n_in,
                              void* d_out, int out_size, void* d_ws,
                              size_t ws_size, hipStream_t stream) {
  const float* depth   = (const float*)d_in[0];
  const float* feature = (const float*)d_in[1];
  const float* w1 = (const float*)d_in[2];
  const float* g1 = (const float*)d_in[3];
  const float* b1 = (const float*)d_in[4];
  const float* m1 = (const float*)d_in[5];
  const float* v1 = (const float*)d_in[6];
  const float* ww = (const float*)d_in[7];
  const float* gw = (const float*)d_in[8];
  const float* bw = (const float*)d_in[9];
  const float* mw = (const float*)d_in[10];
  const float* vw = (const float*)d_in[11];
  const float* wo = (const float*)d_in[12];
  const float* go = (const float*)d_in[13];
  const float* bo = (const float*)d_in[14];
  const float* mo = (const float*)d_in[15];
  const float* vo = (const float*)d_in[16];

  // Workspace layout
  //   cat  : NPIX*128 bf16                = 78,643,200 B  @ 0
  //   Apack: 2*36*32*16 bf16              =     73,728 B  @ 78,643,200
  //   bias : 32 f32                       =        128 B  @ 78,716,928
  //   zrow : 128 bf16 zeros               =        256 B  @ 78,717,056
  //   gacc : NPIX*32 f32                  = 39,321,600 B  @ 78,717,312
  char* ws = (char*)d_ws;
  unsigned short* cat   = (unsigned short*)ws;
  unsigned short* Apack = (unsigned short*)(ws + 78643200);
  float*          bias  = (float*)(ws + 78716928);
  unsigned short* zrow  = (unsigned short*)(ws + 78717056);
  float*          gacc  = (float*)(ws + 78717312);

  static_assert(NPIX % 256 == 0, "pixel tiling");

  (void)hipFuncSetAttribute((const void*)gemm_wmma,
                            hipFuncAttributeMaxDynamicSharedMemorySize, 73728);

  prep_cat<<<NPIX / 32, 256, 0, stream>>>(depth, feature, w1, g1, b1, m1, v1,
                                          cat);
  repack_w<<<(2 * NKB * 32 * 16 + 255) / 256, 256, 0, stream>>>(
      ww, gw, bw, mw, vw, wo, go, bo, mo, vo, Apack, bias,
      (unsigned int*)zrow);
  gemm_wmma<<<NPIX / 256, 256, 73728, stream>>>(cat, Apack, zrow, gacc);
  dkn_sample<<<(NPIX + 255) / 256, 256, 0, stream>>>(gacc, bias, depth,
                                                     (float*)d_out);
}
